// GCN_14370960572608
// MI455X (gfx1250) — compile-verified, matrix operand order checked by reading, not run on hardware
//
#include <hip/hip_runtime.h>

// ---------------------------------------------------------------------------
// GCN forward for MI455X (gfx1250, wave32, WMMA).
//
// Pipeline per layer:  h = H_in @ W  (f16 WMMA, f32 accum)
//                      agg = scatter_add(h[src]*dis[src]*dis[dst] -> dst)
//                      y = agg + h*dis^2 + b   (+ReLU layer 1)
// Head: out = sigmoid(x3 @ Wl + bl)
// ---------------------------------------------------------------------------

typedef __attribute__((ext_vector_type(16))) _Float16 v16h;
typedef __attribute__((ext_vector_type(8)))  _Float16 v8h;
typedef __attribute__((ext_vector_type(8)))  float    v8f;
typedef __attribute__((ext_vector_type(4)))  float    v4f;

// ---------------------------------------------------------------------------
// WMMA GEMM: C[M,N] = A[M,K] (fp32) * B[K,N] (fp32), f16 inputs / f32 accum.
// Block = 256 threads = 8 waves, arranged WR x WC; each wave computes an
// (MI*16) x (NI*16) patch of 16x16 WMMA tiles.  BM = WR*MI*16, BN = WC*NI*16.
//
// Register-pipelined: the next K-tile is fetched from global into VGPRs while
// the current tile (in LDS) feeds the WMMAs; fp32->f16 conversion happens on
// the VGPR->LDS path.  B is fetched as 4-deep k-columns so the transposed
// store into Bs[n][k] is one contiguous ds_store_b64 (2x v_cvt_pk_f16_f32).
// Both WMMA fragment loads are two contiguous ds_load_b128 per lane.
// ANT=true streams A with non-temporal loads (layer 1: x read exactly once).
// ---------------------------------------------------------------------------
template <int WR, int WC, int MI, int NI, bool ANT>
__global__ __launch_bounds__(256) void gemm_wmma_f16(
    const float* __restrict__ A, const float* __restrict__ B,
    float* __restrict__ C, int M, int N, int K)
{
    constexpr int BM = WR * MI * 16;
    constexpr int BN = WC * NI * 16;
    constexpr int BK = 32;
    constexpr int LP = 40;          // LDS pitch in halves: 80B rows, 16B aligned
    constexpr int AV = BM / 32;     // float4 fetches per thread for A tile
    constexpr int BV = BN / 32;     // k-column fetches per thread for B tile

    __shared__ __align__(16) _Float16 As[BM][LP];
    __shared__ __align__(16) _Float16 Bs[BN][LP];

    const int tid    = threadIdx.x;
    const int lane   = tid & 31;
    const int wave   = tid >> 5;
    const int wr     = wave / WC;
    const int wc     = wave % WC;
    const int lm     = lane & 15;   // M/N index inside a 16x16 tile
    const int lh     = lane >> 4;   // K-half selector
    const int blockM = blockIdx.x * BM;
    const int blockN = blockIdx.y * BN;

    const bool fullM = (blockM + BM <= M);
    const bool kvec  = ((K & 3) == 0);   // A row base stays 16B aligned

    v8f acc[MI][NI] = {};
    v4f ra[AV], rb[BV];

    // ---- tile fetch: global -> registers (guards only on edge tiles) -------
    auto fetch = [&](int k0) {
        const bool fastA = fullM && kvec && (k0 + BK <= K);
#pragma unroll
        for (int i = 0; i < AV; ++i) {
            const int idx = i * 256 + tid;
            const int row = idx >> 3;
            const int col = (idx & 7) * 4;
            if (fastA) {
                const v4f* p = (const v4f*)(A + (size_t)(blockM + row) * K + k0 + col);
                ra[i] = ANT ? __builtin_nontemporal_load(p) : *p;
            } else {
                v4f v = {0.f, 0.f, 0.f, 0.f};
                const int gr = blockM + row;
                if (gr < M) {
                    const float* p = A + (size_t)gr * K + k0 + col;
                    if (k0 + col + 0 < K) v.x = p[0];
                    if (k0 + col + 1 < K) v.y = p[1];
                    if (k0 + col + 2 < K) v.z = p[2];
                    if (k0 + col + 3 < K) v.w = p[3];
                }
                ra[i] = v;
            }
        }
        const bool fastB = (k0 + BK <= K);
#pragma unroll
        for (int i = 0; i < BV; ++i) {
            const int idx = i * 256 + tid;
            const int n   = idx % BN;          // column within tile
            const int kq  = (idx / BN) * 4;    // 4-deep k group
            const float* p = B + (size_t)(k0 + kq) * N + blockN + n;
            v4f v = {0.f, 0.f, 0.f, 0.f};
            if (fastB) {
                v.x = p[0];
                v.y = p[N];
                v.z = p[2 * N];
                v.w = p[3 * N];
            } else {
                if (k0 + kq + 0 < K) v.x = p[0];
                if (k0 + kq + 1 < K) v.y = p[N];
                if (k0 + kq + 2 < K) v.z = p[2 * N];
                if (k0 + kq + 3 < K) v.w = p[3 * N];
            }
            rb[i] = v;
        }
    };

    // ---- registers -> LDS (fp32 -> f16 conversion, b64 stores) -------------
    auto stage = [&]() {
#pragma unroll
        for (int i = 0; i < AV; ++i) {
            const int idx = i * 256 + tid;
            const int row = idx >> 3;
            const int col = (idx & 7) * 4;
            _Float16* d = &As[row][col];
            d[0] = (_Float16)ra[i].x;
            d[1] = (_Float16)ra[i].y;
            d[2] = (_Float16)ra[i].z;
            d[3] = (_Float16)ra[i].w;
        }
#pragma unroll
        for (int i = 0; i < BV; ++i) {
            const int idx = i * 256 + tid;
            const int n   = idx % BN;
            const int kq  = (idx / BN) * 4;
            _Float16* d = &Bs[n][kq];          // contiguous in k -> b64 store
            d[0] = (_Float16)rb[i].x;
            d[1] = (_Float16)rb[i].y;
            d[2] = (_Float16)rb[i].z;
            d[3] = (_Float16)rb[i].w;
        }
    };

    fetch(0);
    for (int k0 = 0; k0 < K; k0 += BK) {
        stage();
        __syncthreads();

        if (k0 + BK < K) fetch(k0 + BK);   // overlap next global fetch w/ WMMA

        v16h af[MI], bf[NI];
#pragma unroll
        for (int mi = 0; mi < MI; ++mi) {
            const int row = wr * MI * 16 + mi * 16 + lm;
            const v8h lo = *(const v8h*)&As[row][lh * 8];        // K = lh*8+0..7
            const v8h hi = *(const v8h*)&As[row][16 + lh * 8];   // K = 16+lh*8+0..7
            af[mi] = __builtin_shufflevector(lo, hi, 0, 1, 2, 3, 4, 5, 6, 7,
                                             8, 9, 10, 11, 12, 13, 14, 15);
        }
#pragma unroll
        for (int ni = 0; ni < NI; ++ni) {
            const int col = wc * NI * 16 + ni * 16 + lm;
            const v8h lo = *(const v8h*)&Bs[col][lh * 8];
            const v8h hi = *(const v8h*)&Bs[col][16 + lh * 8];
            bf[ni] = __builtin_shufflevector(lo, hi, 0, 1, 2, 3, 4, 5, 6, 7,
                                             8, 9, 10, 11, 12, 13, 14, 15);
        }
#pragma unroll
        for (int mi = 0; mi < MI; ++mi)
#pragma unroll
            for (int ni = 0; ni < NI; ++ni)
                acc[mi][ni] = __builtin_amdgcn_wmma_f32_16x16x32_f16(
                    false, af[mi], false, bf[ni], (short)0, acc[mi][ni],
                    false, false);
        __syncthreads();
    }

    // ---- store: D layout = VGPR r -> M = r + 8*lh, N = lm ------------------
#pragma unroll
    for (int mi = 0; mi < MI; ++mi) {
#pragma unroll
        for (int ni = 0; ni < NI; ++ni) {
            const int gm = blockM + wr * MI * 16 + mi * 16 + lh * 8;
            const int gn = blockN + wc * NI * 16 + ni * 16 + lm;
#pragma unroll
            for (int r = 0; r < 8; ++r) {
                const int row = gm + r;
                if (row < M) C[(size_t)row * N + gn] = acc[mi][ni][r];
            }
        }
    }
}

// ---------------------------------------------------------------------------
// Graph kernels
// ---------------------------------------------------------------------------
__global__ void zero_f(float* __restrict__ p, long long n) {
    long long t = (long long)blockIdx.x * blockDim.x + threadIdx.x;
    if (t < n) p[t] = 0.f;
}

__global__ void deg_accum(const long long* __restrict__ dst,
                          float* __restrict__ deg, int E) {
    int e = blockIdx.x * blockDim.x + threadIdx.x;
    if (e < E) atomicAdd(&deg[(int)dst[e]], 1.0f);
}

__global__ void to_dis(float* __restrict__ p, int n) {
    int i = blockIdx.x * blockDim.x + threadIdx.x;
    if (i < n) p[i] = rsqrtf(p[i] + 1.0f);  // +1 self-loop
}

template <int F>
__global__ void scatter_edges(const float* __restrict__ h,
                              const long long* __restrict__ src,
                              const long long* __restrict__ dst,
                              const float* __restrict__ dis,
                              float* __restrict__ agg, int E) {
    constexpr int PER = F / 4;
    long long t = (long long)blockIdx.x * blockDim.x + threadIdx.x;
    if (t >= (long long)E * PER) return;
    const int e = (int)(t / PER);
    const int f = (int)(t % PER) * 4;
    const int s = (int)src[e];
    const int d = (int)dst[e];
    const float w = dis[s] * dis[d];
    const float4 hv = *(const float4*)(h + (size_t)s * F + f);
    float* a = agg + (size_t)d * F + f;
    atomicAdd(a + 0, hv.x * w);
    atomicAdd(a + 1, hv.y * w);
    atomicAdd(a + 2, hv.z * w);
    atomicAdd(a + 3, hv.w * w);
}

__global__ void finalize_gcn(const float* __restrict__ agg,
                             const float* __restrict__ h,
                             const float* __restrict__ dis,
                             const float* __restrict__ bias,
                             float* __restrict__ out, int Mn, int F,
                             int do_relu) {
    long long t = (long long)blockIdx.x * blockDim.x + threadIdx.x;
    if (t >= (long long)Mn * F) return;
    const int i = (int)(t / F);
    const int f = (int)(t % F);
    const float di = dis[i];
    float v = agg[t] + h[t] * di * di + bias[f];
    if (do_relu) v = fmaxf(v, 0.f);
    out[t] = v;
}

__global__ void head_kernel(const float* __restrict__ x3,
                            const float* __restrict__ Wl,
                            const float* __restrict__ bl,
                            float* __restrict__ out, int Mn) {
    int i = blockIdx.x * blockDim.x + threadIdx.x;
    if (i >= Mn) return;
    float s = bl[0];
    const float* row = x3 + (size_t)i * 64;
#pragma unroll
    for (int f = 0; f < 64; f += 4) {
        const float4 v = *(const float4*)(row + f);
        const float4 w = *(const float4*)(Wl + f);
        s += v.x * w.x + v.y * w.y + v.z * w.z + v.w * w.w;
    }
    out[i] = 1.0f / (1.0f + __expf(-s));
}

static inline unsigned cdivu(long long a, long long b) {
    return (unsigned)((a + b - 1) / b);
}

extern "C" void kernel_launch(void* const* d_in, const int* in_sizes, int n_in,
                              void* d_out, int out_size, void* d_ws,
                              size_t ws_size, hipStream_t stream) {
    const float*     x   = (const float*)d_in[0];
    const long long* ei  = (const long long*)d_in[1];
    const float*     W1  = (const float*)d_in[2];
    const float*     b1  = (const float*)d_in[3];
    const float*     W2  = (const float*)d_in[4];
    const float*     b2  = (const float*)d_in[5];
    const float*     W3  = (const float*)d_in[6];
    const float*     b3  = (const float*)d_in[7];
    const float*     Wl  = (const float*)d_in[8];
    const float*     bl  = (const float*)d_in[9];

    const int D0 = 3000, D1 = 256, D2 = 128, D3 = 64;
    const int Nn = in_sizes[0] / D0;   // 50000 nodes
    const int E  = in_sizes[1] / 2;    // 1.6M edges
    const long long* src = ei;
    const long long* dst = ei + E;

    // workspace layout (floats): dis | buf1 (Nn*256) | buf2 (Nn*256)  ~103 MB
    float* ws   = (float*)d_ws;
    float* dis  = ws;
    float* buf1 = dis + Nn;
    float* buf2 = buf1 + (size_t)Nn * D1;

    float* out = (float*)d_out;       // [Nn] sigmoid outputs
    float* x3  = out + Nn;            // [Nn*64] third-layer activations

    // ---- degrees & symmetric norm -----------------------------------------
    zero_f<<<cdivu(Nn, 256), 256, 0, stream>>>(dis, Nn);
    deg_accum<<<cdivu(E, 256), 256, 0, stream>>>(dst, dis, E);
    to_dis<<<cdivu(Nn, 256), 256, 0, stream>>>(dis, Nn);

    // ---- layer 1: h1 = x @ W1  (BM=64, BN=256: x streamed from HBM once,
    //      non-temporal so the 600MB stream never thrashes L2) --------------
    gemm_wmma_f16<2, 4, 2, 4, true>
        <<<dim3(cdivu(Nn, 64), D1 / 256), 256, 0, stream>>>(x, W1, buf1, Nn, D1, D0);
    zero_f<<<cdivu((long long)Nn * D1, 256), 256, 0, stream>>>(buf2, (long long)Nn * D1);
    scatter_edges<256><<<cdivu((long long)E * 64, 256), 256, 0, stream>>>(
        buf1, src, dst, dis, buf2, E);
    finalize_gcn<<<cdivu((long long)Nn * D1, 256), 256, 0, stream>>>(
        buf2, buf1, dis, b1, buf2, Nn, D1, 1);

    // ---- layer 2: h2 = y1 @ W2  (BM=128, BN=128; y1 is L2-resident) -------
    gemm_wmma_f16<4, 2, 2, 4, false>
        <<<dim3(cdivu(Nn, 128), D2 / 128), 256, 0, stream>>>(buf2, W2, buf1, Nn, D2, D1);
    zero_f<<<cdivu((long long)Nn * D2, 256), 256, 0, stream>>>(buf2, (long long)Nn * D2);
    scatter_edges<128><<<cdivu((long long)E * 32, 256), 256, 0, stream>>>(
        buf1, src, dst, dis, buf2, E);
    finalize_gcn<<<cdivu((long long)Nn * D2, 256), 256, 0, stream>>>(
        buf2, buf1, dis, b2, buf2, Nn, D2, 0);

    // ---- layer 3: h3 = y2 @ W3  (BM=128, BN=64) ---------------------------
    gemm_wmma_f16<4, 2, 2, 2, false>
        <<<dim3(cdivu(Nn, 128), D3 / 64), 256, 0, stream>>>(buf2, W3, buf1, Nn, D3, D2);
    zero_f<<<cdivu((long long)Nn * D3, 256), 256, 0, stream>>>(buf2, (long long)Nn * D3);
    scatter_edges<64><<<cdivu((long long)E * 16, 256), 256, 0, stream>>>(
        buf1, src, dst, dis, buf2, E);
    finalize_gcn<<<cdivu((long long)Nn * D3, 256), 256, 0, stream>>>(
        buf2, buf1, dis, b3, x3, Nn, D3, 0);  // x3 -> d_out[Nn:]

    // ---- head: sigmoid(x3 @ Wl + bl) -> d_out[0:Nn] -----------------------
    head_kernel<<<cdivu(Nn, 256), 256, 0, stream>>>(x3, Wl, bl, out, Nn);
}
